// TwoClassOnePosMaskedLoss_43679817400367
// MI455X (gfx1250) — matrix operand order repaired and checked by reading, not compiled, not run on hardware
//
#include <hip/hip_runtime.h>
#include <hip/hip_bf16.h>
#include <math.h>

typedef __attribute__((ext_vector_type(16))) _Float16 v16h;
typedef __attribute__((ext_vector_type(8)))  _Float16 v8h;
typedef __attribute__((ext_vector_type(4)))  _Float16 v4h;
typedef __attribute__((ext_vector_type(8)))  float    v8f;
typedef __attribute__((ext_vector_type(4)))  float    v4f;

#define B1N  4096
#define BTOT 8192
#define FD   128
#define NSLICE   4
#define SLICE_W  2048

// Work in base-2 exp/log domain: v = sim*(1/(TAU*ln2)).
// MFIX = upper bound of v for unit-norm rows -> fixed softmax shift, args to
// v_exp_f32 are in [-41.3, 0]: no libm range fixup needed.
__constant__ constexpr float SCALE_B2 = 20.6158809149f;  // 1/(0.07*ln2)
__constant__ constexpr float MFIX     = 20.6158809149f;
__constant__ constexpr float LN2F     = 0.69314718056f;
__constant__ constexpr float NEGBIG   = -3.0e38f;

// ---------------- Kernel 1: concat + L2-normalize + fp32->fp16 ----------------
__global__ void __launch_bounds__(256)
normalize_f16_kernel(const float* __restrict__ f1, const float* __restrict__ f2,
                     _Float16* __restrict__ out) {
  const int gtid = blockIdx.x * blockDim.x + threadIdx.x;
  const int row  = gtid >> 5;
  const int lane = gtid & 31;
  if (row >= BTOT) return;
  const float* src = (row < B1N) ? (f1 + (size_t)row * FD)
                                 : (f2 + (size_t)(row - B1N) * FD);
  v4f x = *(const v4f*)(src + lane * 4);
  float ss = x.x * x.x + x.y * x.y + x.z * x.z + x.w * x.w;
#pragma unroll
  for (int m = 16; m > 0; m >>= 1) ss += __shfl_xor(ss, m, 32);
  const float inv = 1.0f / fmaxf(sqrtf(ss), 1e-12f);
  v4h h;
  h.x = (_Float16)(x.x * inv);
  h.y = (_Float16)(x.y * inv);
  h.z = (_Float16)(x.z * inv);
  h.w = (_Float16)(x.w * inv);
  *(v4h*)(out + (size_t)row * FD + lane * 4) = h;
}

// ---------------- Kernel 2: WMMA GEMM slice + fused row partials ----------------
__device__ __forceinline__ void load_bfr(const _Float16* __restrict__ feats,
                                         int nb, int ln16, int hg, v16h b[4]) {
  const _Float16* brow = feats + (size_t)(nb + ln16) * FD;
#pragma unroll
  for (int kc = 0; kc < 4; ++kc) b[kc] = *(const v16h*)(brow + kc * 32 + 16 * hg);
}

__device__ __forceinline__ v8f wmma_chain(const v16h a[4], const v16h b[4]) {
  v8f c = {0.0f, 0.0f, 0.0f, 0.0f, 0.0f, 0.0f, 0.0f, 0.0f};
#pragma unroll
  for (int kc = 0; kc < 4; ++kc)
    c = __builtin_amdgcn_wmma_f32_16x16x32_f16(false, a[kc], false, b[kc],
                                               (short)0, c, false, false);
  return c;
}

// Wave = (rowTile, slice): rows [16*rowTile, +16), cols [2048*slice, +2048).
// Blocks pack 4 consecutive rowTiles at the same slice -> B tiles reuse via L0.
// A slice is entirely same-class or entirely other-class (2048 | 4096).
__global__ void __launch_bounds__(128)
simloss_kernel(const _Float16* __restrict__ feats,
               float* __restrict__ msameP,    // [2][BTOT] raw same-class max
               float* __restrict__ sP) {      // [2][BTOT] fixed-shift exp sums
  const int tid   = threadIdx.x;
  const int lane  = tid & 31;
  const int hg    = lane >> 4;
  const int ln16  = lane & 15;
  const int slice   = __builtin_amdgcn_readfirstlane((int)(blockIdx.x >> 7));
  const int rowTile = __builtin_amdgcn_readfirstlane(
                          (int)((blockIdx.x & 127) << 2) | (tid >> 5));
  const int m_base  = rowTile << 4;
  const int nbLo    = slice * SLICE_W;
  const int nbHi    = nbLo + SLICE_W;
  const bool sameSlice = ((nbLo < B1N) == (m_base < B1N));  // scalar
  const int  sIdx      = slice & 1;                          // 0/1 within class half
  const int  myrow0    = m_base + 8 * hg;                    // row of c[r] = myrow0+r

  // Pin A fragments (16 rows x K=128 -> 4 chunks of 32).
  const _Float16* arow = feats + (size_t)(m_base + ln16) * FD;
  v16h afr[4];
#pragma unroll
  for (int kc = 0; kc < 4; ++kc) {
    afr[kc] = *(const v16h*)(arow + kc * 32 + 16 * hg);
  }

  v16h b0[4], b1[4];
  load_bfr(feats, nbLo, ln16, hg, b0);
  float acc[8];

  if (sameSlice) {
    // ---- same-class slice: raw running max (scale applied in finalize) ----
#pragma unroll
    for (int r = 0; r < 8; ++r) acc[r] = NEGBIG;
    for (int nb = nbLo; nb < nbHi; nb += 32) {
      load_bfr(feats, nb + 16, ln16, hg, b1);          // prefetch half B
      {
        v8f c = wmma_chain(afr, b0);
        if (nb == m_base) {                             // uniform: diagonal tile
#pragma unroll
          for (int r = 0; r < 8; ++r) {
            float v = (nb + ln16 == myrow0 + r) ? NEGBIG : c[r];
            acc[r] = fmaxf(acc[r], v);
          }
        } else {
#pragma unroll
          for (int r = 0; r < 8; ++r) acc[r] = fmaxf(acc[r], c[r]);
        }
      }
      if (nb + 32 < nbHi) load_bfr(feats, nb + 32, ln16, hg, b0);
      {
        v8f c = wmma_chain(afr, b1);
        if (nb + 16 == m_base) {
#pragma unroll
          for (int r = 0; r < 8; ++r) {
            float v = (nb + 16 + ln16 == myrow0 + r) ? NEGBIG : c[r];
            acc[r] = fmaxf(acc[r], v);
          }
        } else {
#pragma unroll
          for (int r = 0; r < 8; ++r) acc[r] = fmaxf(acc[r], c[r]);
        }
      }
    }
#pragma unroll
    for (int r = 0; r < 8; ++r)
#pragma unroll
      for (int m = 1; m < 16; m <<= 1)
        acc[r] = fmaxf(acc[r], __shfl_xor(acc[r], m, 32));
    if (ln16 == 0) {
      float* dst = msameP + sIdx * BTOT + myrow0;
      v4f a = {acc[0], acc[1], acc[2], acc[3]};
      v4f b = {acc[4], acc[5], acc[6], acc[7]};
      *(v4f*)(dst) = a;
      *(v4f*)(dst + 4) = b;
    }
  } else {
    // ---- other-class slice: fixed-shift exp accumulation ----
#pragma unroll
    for (int r = 0; r < 8; ++r) acc[r] = 0.0f;
    for (int nb = nbLo; nb < nbHi; nb += 32) {
      load_bfr(feats, nb + 16, ln16, hg, b1);
      {
        v8f c = wmma_chain(afr, b0);
#pragma unroll
        for (int r = 0; r < 8; ++r)
          acc[r] += __builtin_amdgcn_exp2f(fmaf(c[r], SCALE_B2, -MFIX));
      }
      if (nb + 32 < nbHi) load_bfr(feats, nb + 32, ln16, hg, b0);
      {
        v8f c = wmma_chain(afr, b1);
#pragma unroll
        for (int r = 0; r < 8; ++r)
          acc[r] += __builtin_amdgcn_exp2f(fmaf(c[r], SCALE_B2, -MFIX));
      }
    }
#pragma unroll
    for (int r = 0; r < 8; ++r)
#pragma unroll
      for (int m = 1; m < 16; m <<= 1)
        acc[r] += __shfl_xor(acc[r], m, 32);
    if (ln16 == 0) {
      float* dst = sP + sIdx * BTOT + myrow0;
      v4f a = {acc[0], acc[1], acc[2], acc[3]};
      v4f b = {acc[4], acc[5], acc[6], acc[7]};
      *(v4f*)(dst) = a;
      *(v4f*)(dst + 4) = b;
    }
  }
}

// ---------------- Kernel 3: merge slices, per-row loss, mean ----------------
__global__ void __launch_bounds__(1024)
finalize_kernel(const float* __restrict__ msameP, const float* __restrict__ sP,
                float* __restrict__ out) {
  __shared__ float sm[1024];
  const int t = threadIdx.x;
  float accum = 0.0f;
#pragma unroll
  for (int i = t; i < BTOT; i += 1024) {
    const float ms = fmaxf(msameP[i], msameP[BTOT + i]) * SCALE_B2;  // scaled pos
    const float S  = sP[i] + sP[BTOT + i];
    const float sf = S + __builtin_amdgcn_exp2f(ms - MFIX);
    accum += LN2F * (__builtin_amdgcn_logf(sf) + MFIX - ms);
  }
  sm[t] = accum;
  __syncthreads();
#pragma unroll
  for (int s = 512; s > 0; s >>= 1) {
    if (t < s) sm[t] += sm[t + s];
    __syncthreads();
  }
  if (t == 0) out[0] = sm[0] * (1.0f / (float)BTOT);
}

extern "C" void kernel_launch(void* const* d_in, const int* in_sizes, int n_in,
                              void* d_out, int out_size, void* d_ws, size_t ws_size,
                              hipStream_t stream) {
  const float* f1 = (const float*)d_in[0];
  const float* f2 = (const float*)d_in[1];
  float* out = (float*)d_out;

  _Float16* feats_h = (_Float16*)d_ws;                         // 2 MB
  float* msameP = (float*)((char*)d_ws + (size_t)BTOT * FD * sizeof(_Float16));
  float* sP     = msameP + 2 * BTOT;                           // 2*[2][8192] floats

  normalize_f16_kernel<<<1024, 256, 0, stream>>>(f1, f2, feats_h);
  // 2048 waves: 512 rowTiles x 4 column slices; 4 waves/block -> 512 blocks
  simloss_kernel<<<512, 128, 0, stream>>>(feats_h, msameP, sP);
  finalize_kernel<<<1, 1024, 0, stream>>>(msameP, sP, out);
}
